// GroupedQueryAttention_23957327577818
// MI455X (gfx1250) — matrix operand order repaired and checked by reading, not compiled, not run on hardware
//
#include <hip/hip_runtime.h>
#include <hip/hip_bf16.h>

// ---------------------------------------------------------------------------
// GQA attention block for MI455X (gfx1250, wave32, WMMA 16x16x32 f16/f32-acc)
// f16 data plane + async global->LDS tile staging (ASYNCcnt).
// N_EMBD=2048, N_HEAD=16, N_KV_HEAD=4, HEAD_DIM=128, B=4, S=1024 -> M=B*S=4096
// ---------------------------------------------------------------------------

typedef _Float16 v16h __attribute__((ext_vector_type(16)));
typedef _Float16 h8   __attribute__((ext_vector_type(8)));
typedef _Float16 h4   __attribute__((ext_vector_type(4)));
typedef float    v8f  __attribute__((ext_vector_type(8)));
typedef float    f4v  __attribute__((ext_vector_type(4)));
typedef int      i4v  __attribute__((vector_size(4 * sizeof(int))));  // b128 payload

#define TOK   4096     // B*S
#define SLEN  1024
#define EMB   2048
#define HDIM  128
#define KVDIM 512      // N_KV_HEAD * HDIM

// ---------------- async global -> LDS (CDNA5) with safe fallback -----------
#if __has_builtin(__builtin_amdgcn_global_load_async_to_lds_b128) && \
    __has_builtin(__builtin_amdgcn_s_wait_asynccnt)
#define USE_ASYNC_LDS 1
#else
#define USE_ASYNC_LDS 0
#endif

__device__ __forceinline__ void lds_cp16(void* s, const void* g) {
#if USE_ASYNC_LDS
  // generic->AS1 is bit-identical for global; generic LDS addr truncates to
  // the LDS offset (aperture rule), so integer casts are semantically exact.
  __builtin_amdgcn_global_load_async_to_lds_b128(
      (i4v __attribute__((address_space(1)))*)(unsigned long long)g,
      (i4v __attribute__((address_space(3)))*)(unsigned long long)s,
      0, 0);
#else
  *(h8*)s = *(const h8*)g;
#endif
}
__device__ __forceinline__ void lds_cp_wait() {
#if USE_ASYNC_LDS
  __builtin_amdgcn_s_wait_asynccnt(0);
#endif
}

// ---------------------------- small helpers --------------------------------
__device__ __forceinline__ v8f z8() {
  v8f z;
#pragma unroll
  for (int i = 0; i < 8; ++i) z[i] = 0.0f;
  return z;
}

// load 8 fp32 (16B-aligned) and convert to 8 f16
__device__ __forceinline__ h8 cvt8(const float* p) {
  f4v a = *(const f4v*)p;
  f4v b = *(const f4v*)(p + 4);
  h8 r;
  r[0] = (_Float16)a[0]; r[1] = (_Float16)a[1];
  r[2] = (_Float16)a[2]; r[3] = (_Float16)a[3];
  r[4] = (_Float16)b[0]; r[5] = (_Float16)b[1];
  r[6] = (_Float16)b[2]; r[7] = (_Float16)b[3];
  return r;
}

__device__ __forceinline__ v16h mk16(h8 lo, h8 hi) {
  return __builtin_shufflevector(lo, hi, 0,1,2,3,4,5,6,7,8,9,10,11,12,13,14,15);
}

__device__ __forceinline__ v8f wmma16(v16h a, v16h b, v8f c) {
  // D = A(16x32 f16) * B(32x16 f16) + C(16x16 f32)
  return __builtin_amdgcn_wmma_f32_16x16x32_f16(
      false, a, false, b, (short)0, c, false, false);
}

__device__ __forceinline__ float rmax16(float v) {
  v = fmaxf(v, __shfl_xor(v, 1, 32));
  v = fmaxf(v, __shfl_xor(v, 2, 32));
  v = fmaxf(v, __shfl_xor(v, 4, 32));
  v = fmaxf(v, __shfl_xor(v, 8, 32));
  return v;
}
__device__ __forceinline__ float rsum16(float v) {
  v += __shfl_xor(v, 1, 32);
  v += __shfl_xor(v, 2, 32);
  v += __shfl_xor(v, 4, 32);
  v += __shfl_xor(v, 8, 32);
  return v;
}
__device__ __forceinline__ float rsum32(float v) {
  v = rsum16(v);
  v += __shfl_xor(v, 16, 32);
  return v;
}

// ---------------------------------------------------------------------------
// Kernel 0: fp32 -> f16 bulk convert. Each thread converts 8 elements.
// ---------------------------------------------------------------------------
__global__ void __launch_bounds__(256)
cvt_f16(const float* __restrict__ in, _Float16* __restrict__ out, int n8) {
  const int i = blockIdx.x * 256 + threadIdx.x;
  if (i < n8)
    *(h8*)(out + (size_t)i * 8) = cvt8(in + (size_t)i * 8);
}

// ---------------------------------------------------------------------------
// Kernel 1/4: C[m][n] = sum_k A[m][k] * W[n][k]   (A,W f16 row-major, K-contig)
// WG = 256 threads (8 waves). Tile: 128(M) x 128(N), K-step 32.
// Wave (wm in 0..3, wn in 0..1) computes 32x64 as 2x4 WMMA tiles.
// Tiles staged via async global->LDS b128 copies (no convert needed).
// ---------------------------------------------------------------------------
template <typename OutT>
__global__ void __launch_bounds__(256)
gemm_nt_f16(const _Float16* __restrict__ A, const _Float16* __restrict__ W,
            OutT* __restrict__ C, int M, int N, int K) {
  __shared__ _Float16 As[128][40];   // 80B row stride (16B multiple)
  __shared__ _Float16 Ws[128][40];

  const int t    = threadIdx.x;
  const int lane = t & 31, wvid = t >> 5;
  const int wm = wvid & 3, wn = wvid >> 2;
  const int l  = lane & 15, hg = lane >> 4;
  const int m0 = blockIdx.y * 128, n0 = blockIdx.x * 128;

  v8f acc[2][4];
#pragma unroll
  for (int i = 0; i < 2; ++i)
#pragma unroll
    for (int j = 0; j < 4; ++j) acc[i][j] = z8();

  for (int kb = 0; kb < K; kb += 32) {
    __syncthreads();
    // 128 rows x 4 chunks(8 halfs) = 512 chunks each for A and W
#pragma unroll
    for (int cc = t, it = 0; it < 2; ++it, cc += 256) {
      const int r = cc >> 2, co = (cc & 3) * 8;
      const _Float16* ap = A + (size_t)(m0 + r) * K + kb + co;
      const _Float16* wp = W + (size_t)(n0 + r) * K + kb + co;
      lds_cp16(&As[r][co], ap);
      lds_cp16(&Ws[r][co], wp);
      __builtin_prefetch(ap + 32, 0, 1);   // global_prefetch_b8: next K tile
      __builtin_prefetch(wp + 32, 0, 1);
    }
    lds_cp_wait();
    __syncthreads();

#pragma unroll
    for (int i = 0; i < 2; ++i) {
      const int arow = wm * 32 + i * 16 + l;
      // A frag: lanes 0-15 K=[0..7],[16..23]; lanes 16-31 K=[8..15],[24..31]
      v16h af = mk16(*(const h8*)&As[arow][hg * 8],
                     *(const h8*)&As[arow][16 + hg * 8]);
#pragma unroll
      for (int j = 0; j < 4; ++j) {
        const int brow = wn * 64 + j * 16 + l;
        // B frag: lanes 0-15 K=[0..7],[16..23]; lanes 16-31 K=[8..15],[24..31]
        v16h bf = mk16(*(const h8*)&Ws[brow][hg * 8],
                       *(const h8*)&Ws[brow][16 + hg * 8]);
        acc[i][j] = wmma16(af, bf, acc[i][j]);
      }
    }
  }

#pragma unroll
  for (int i = 0; i < 2; ++i)
#pragma unroll
    for (int j = 0; j < 4; ++j) {
      const int row = m0 + wm * 32 + i * 16 + 8 * hg;   // + r below
      const int col = n0 + wn * 64 + j * 16 + l;
      OutT* cp = C + (size_t)row * N + col;
#pragma unroll
      for (int r = 0; r < 8; ++r) cp[(size_t)r * N] = (OutT)acc[i][j][r];
    }
}

// ---------------------------------------------------------------------------
// Kernel 2: per-(token, head) RMSNorm over HDIM then RoPE, in-place on f16.
// One wave per (token, head); fp32 math. X layout: [token][head][128].
// ---------------------------------------------------------------------------
__global__ void __launch_bounds__(128)
rmsrope(_Float16* __restrict__ X, const float* __restrict__ NW, int nheads) {
  const int wid   = blockIdx.x * 4 + (threadIdx.x >> 5);
  const int lane  = threadIdx.x & 31;
  const int token = wid / nheads;
  const int head  = wid - token * nheads;
  const int pos   = token & (SLEN - 1);

  _Float16* p = X + (size_t)token * nheads * HDIM + head * HDIM + lane * 4;
  h4 xh = *(h4*)p;
  float x0 = (float)xh[0], x1 = (float)xh[1], x2 = (float)xh[2], x3 = (float)xh[3];
  float ss = x0 * x0 + x1 * x1 + x2 * x2 + x3 * x3;
  ss = rsum32(ss);
  const float inv = rsqrtf(ss * (1.0f / 128.0f) + 1e-6f);

  const float* wp = NW + lane * 4;
  float n0 = x0 * inv * wp[0];
  float n1 = x1 * inv * wp[1];
  float n2 = x2 * inv * wp[2];
  float n3 = x3 * inv * wp[3];

  // pair indices 2*lane and 2*lane+1; inv_freq = 10000^(-i/64)
  const float kln = 0.14391156831212787f;           // ln(10000)/64
  const float a0 = (float)pos * __expf(-kln * (float)(2 * lane));
  const float a1 = (float)pos * __expf(-kln * (float)(2 * lane + 1));
  float s0, c0, s1, c1;
  __sincosf(a0, &s0, &c0);
  __sincosf(a1, &s1, &c1);

  h4 r;
  r[0] = (_Float16)(n0 * c0 - n1 * s0);
  r[1] = (_Float16)(n0 * s0 + n1 * c0);
  r[2] = (_Float16)(n2 * c1 - n3 * s1);
  r[3] = (_Float16)(n2 * s1 + n3 * c1);
  *(h4*)p = r;
}

// ---------------------------------------------------------------------------
// Kernel 3: causal flash attention, GQA 4:1, f16 Q/K/V, fp32 softmax state.
// Grid: (S/64, B*H). WG = 128 thr = 4 waves; wave w owns Q rows [qb+16w, +16).
// K tile staged via async global->LDS; V staged transposed (sync).
// Fully-masked KV blocks are exact no-ops -> uniform control flow.
// All 8-trip row loops force-unrolled: static VGPR indexing (no v_movrels).
// ---------------------------------------------------------------------------
__global__ void __launch_bounds__(128)
attn_fwd(const _Float16* __restrict__ Q, const _Float16* __restrict__ Kc,
         const _Float16* __restrict__ Vc, _Float16* __restrict__ O) {
  __shared__ _Float16 Kt[32][136];        // [kv row][d]   272B stride
  __shared__ _Float16 Vt[HDIM][40];       // [d][kv row]   transposed
  __shared__ _Float16 Ps[4][16][40];      // per-wave P re-layout patch

  const int t = threadIdx.x, lane = t & 31, w = t >> 5;
  const int l = lane & 15, hg = lane >> 4;
  const int bh = blockIdx.y, b = bh >> 4, h = bh & 15, kvh = h >> 2;
  const int qblock = blockIdx.x * 64;
  const int qbase  = qblock + w * 16;

  // Q fragments for this wave's 16 rows, straight from global f16.
  v16h qf[4];
  const _Float16* qp = Q + (size_t)(b * SLEN + qbase + l) * EMB + h * HDIM;
#pragma unroll
  for (int kk = 0; kk < 4; ++kk)
    qf[kk] = mk16(*(const h8*)(qp + kk * 32 + hg * 8),
                  *(const h8*)(qp + kk * 32 + 16 + hg * 8));

  v8f o[8];
#pragma unroll
  for (int jo = 0; jo < 8; ++jo) o[jo] = z8();
  float mrun[8], lrun[8];
#pragma unroll
  for (int r = 0; r < 8; ++r) { mrun[r] = -1e30f; lrun[r] = 0.0f; }

  const float scale = 0.08838834764831845f;   // 1/sqrt(128)
  const int kend = qblock + 64;               // causal KV extent for this block
  const int krr = t >> 2, kcc = (t & 3) * 32; // V transpose mapping

  for (int kb = 0; kb < kend; kb += 32) {
    __syncthreads();
    // K tile: 32 rows x 16 chunks(8 halfs) = 512 chunks, async b128 copies
#pragma unroll
    for (int cc = t, it = 0; it < 4; ++it, cc += 128) {
      const int r = cc >> 4, co = (cc & 15) * 8;
      lds_cp16(&Kt[r][co],
               Kc + (size_t)(b * SLEN + kb + r) * KVDIM + kvh * HDIM + co);
    }
    // V tile transposed: thread -> (row krr, 32 cols at kcc)
    const _Float16* vp =
        Vc + (size_t)(b * SLEN + kb + krr) * KVDIM + kvh * HDIM + kcc;
#pragma unroll
    for (int c = 0; c < 32; c += 8) {
      h8 vv = *(const h8*)(vp + c);
#pragma unroll
      for (int e = 0; e < 8; ++e) Vt[kcc + c + e][krr] = vv[e];
    }
    lds_cp_wait();
    __syncthreads();

    // scores: two 16x16 C tiles (cols kb..kb+15, kb+16..kb+31)
    v8f c0 = z8(), c1 = z8();
#pragma unroll
    for (int kk = 0; kk < 4; ++kk) {
      v16h b0 = mk16(*(const h8*)&Kt[l][kk * 32 + hg * 8],
                     *(const h8*)&Kt[l][kk * 32 + 16 + hg * 8]);
      v16h b1 = mk16(*(const h8*)&Kt[16 + l][kk * 32 + hg * 8],
                     *(const h8*)&Kt[16 + l][kk * 32 + 16 + hg * 8]);
      c0 = wmma16(qf[kk], b0, c0);
      c1 = wmma16(qf[kk], b1, c1);
    }

    // online softmax (row = qbase + r + 8*hg lives across lanes of this half)
    float alpha[8];
#pragma unroll
    for (int r = 0; r < 8; ++r) {
      const int row = qbase + r + 8 * hg;
      float s0 = c0[r] * scale, s1 = c1[r] * scale;
      if (kb + l > row)      s0 = -1e30f;
      if (kb + 16 + l > row) s1 = -1e30f;
      float mx = rmax16(fmaxf(s0, s1));
      float mn = fmaxf(mrun[r], mx);
      float a  = __expf(mrun[r] - mn);
      float p0 = __expf(s0 - mn);
      float p1 = __expf(s1 - mn);
      lrun[r] = lrun[r] * a + rsum16(p0 + p1);
      mrun[r] = mn;
      alpha[r] = a;
      Ps[w][r + 8 * hg][l]      = (_Float16)p0;
      Ps[w][r + 8 * hg][16 + l] = (_Float16)p1;
    }
#pragma unroll
    for (int jo = 0; jo < 8; ++jo)
#pragma unroll
      for (int r = 0; r < 8; ++r) o[jo][r] *= alpha[r];

    __syncthreads();   // Ps visibility + DS ordering (uniform)

    // PV: A = P (16x32), B = V^T-staged (32 x 128 as 8 N-tiles)
    v16h ap = mk16(*(const h8*)&Ps[w][l][hg * 8],
                   *(const h8*)&Ps[w][l][16 + hg * 8]);
#pragma unroll
    for (int jo = 0; jo < 8; ++jo) {
      v16h bv = mk16(*(const h8*)&Vt[jo * 16 + l][hg * 8],
                     *(const h8*)&Vt[jo * 16 + l][16 + hg * 8]);
      o[jo] = wmma16(ap, bv, o[jo]);
    }
  }

  // epilogue: normalize by l, store f16 at [token][h][d]
  _Float16* op = O + (size_t)(b * SLEN + qbase + 8 * hg) * EMB + h * HDIM;
#pragma unroll
  for (int r = 0; r < 8; ++r) {
    const float inv = 1.0f / lrun[r];
#pragma unroll
    for (int jo = 0; jo < 8; ++jo)
      op[(size_t)r * EMB + jo * 16 + l] = (_Float16)(o[jo][r] * inv);
  }
}

// ---------------------------------------------------------------------------
// Host-side launcher
// ---------------------------------------------------------------------------
extern "C" void kernel_launch(void* const* d_in, const int* in_sizes, int n_in,
                              void* d_out, int out_size, void* d_ws, size_t ws_size,
                              hipStream_t stream) {
  (void)in_sizes; (void)n_in; (void)out_size; (void)ws_size;
  const float* x   = (const float*)d_in[0];
  const float* wq  = (const float*)d_in[1];
  const float* wk  = (const float*)d_in[2];
  const float* wv  = (const float*)d_in[3];
  const float* wo  = (const float*)d_in[4];
  const float* qnw = (const float*)d_in[5];
  const float* knw = (const float*)d_in[6];
  float* out = (float*)d_out;

  // f16 workspace layout
  _Float16* xh  = (_Float16*)d_ws;                         // 4096*2048
  _Float16* wqh = xh  + (size_t)TOK * EMB;                 // 2048*2048
  _Float16* wkh = wqh + (size_t)EMB * EMB;                 // 512*2048
  _Float16* wvh = wkh + (size_t)KVDIM * EMB;               // 512*2048
  _Float16* woh = wvh + (size_t)KVDIM * EMB;               // 2048*2048
  _Float16* qh  = woh + (size_t)EMB * EMB;                 // 4096*2048
  _Float16* kh  = qh  + (size_t)TOK * EMB;                 // 4096*512
  _Float16* vh  = kh  + (size_t)TOK * KVDIM;               // 4096*512
  _Float16* ah  = vh  + (size_t)TOK * KVDIM;               // 4096*2048

  // one-time fp32 -> f16 conversion of activations and weights
  cvt_f16<<<(TOK * EMB)   / (8 * 256), 256, 0, stream>>>(x,  xh,  (TOK * EMB) / 8);
  cvt_f16<<<(EMB * EMB)   / (8 * 256), 256, 0, stream>>>(wq, wqh, (EMB * EMB) / 8);
  cvt_f16<<<(KVDIM * EMB) / (8 * 256), 256, 0, stream>>>(wk, wkh, (KVDIM * EMB) / 8);
  cvt_f16<<<(KVDIM * EMB) / (8 * 256), 256, 0, stream>>>(wv, wvh, (KVDIM * EMB) / 8);
  cvt_f16<<<(EMB * EMB)   / (8 * 256), 256, 0, stream>>>(wo, woh, (EMB * EMB) / 8);

  // QKV projections (A . W^T), f16 out
  gemm_nt_f16<_Float16><<<dim3(EMB / 128,   TOK / 128), 256, 0, stream>>>(xh, wqh, qh, TOK, EMB,   EMB);
  gemm_nt_f16<_Float16><<<dim3(KVDIM / 128, TOK / 128), 256, 0, stream>>>(xh, wkh, kh, TOK, KVDIM, EMB);
  gemm_nt_f16<_Float16><<<dim3(KVDIM / 128, TOK / 128), 256, 0, stream>>>(xh, wvh, vh, TOK, KVDIM, EMB);

  // RMSNorm + RoPE on q (16 heads) and k (4 heads): one wave per (token, head)
  rmsrope<<<(TOK * 16) / 4, 128, 0, stream>>>(qh, qnw, 16);
  rmsrope<<<(TOK * 4)  / 4, 128, 0, stream>>>(kh, knw, 4);

  // causal GQA flash attention
  attn_fwd<<<dim3(SLEN / 64, 4 * 16), 128, 0, stream>>>(qh, kh, vh, ah);

  // output projection, fp32 out
  gemm_nt_f16<float><<<dim3(EMB / 128, TOK / 128), 256, 0, stream>>>(ah, woh, out, TOK, EMB, EMB);
}
